// MyRNN_2439541424712
// MI455X (gfx1250) — matrix operand order repaired and checked by reading, not compiled, not run on hardware
//
#include <hip/hip_runtime.h>
#include <hip/hip_bf16.h>

// Fused tanh-RNN scan for MI455X (gfx1250, wave32, WMMA).
// One block per 16 batch rows; 8 waves each own 32 of the 256 hidden columns.
// Weights live in per-wave f16 register fragments; h ping-pongs through LDS.
// A-fragments are software-pipelined (2 deep) so ds_load_b128 latency overlaps
// the dependent v_wmma chain instead of draining on s_wait_dscnt 0.

typedef __attribute__((ext_vector_type(16))) _Float16 v16h;
typedef __attribute__((ext_vector_type(8)))  _Float16 v8h;
typedef __attribute__((ext_vector_type(4)))  _Float16 v4h;
typedef __attribute__((ext_vector_type(8)))  float    v8f;
typedef __attribute__((ext_vector_type(4)))  float    v4f;

#define RNN_B 2048
#define RNN_S 1024
#define RNN_I 64
#define RNN_H 256
#define RNN_O 32

// LDS row strides (halfs), padded so 16 rows map to 16 distinct 4-bank groups
#define H_STRIDE 264   // 256 + 8
#define X_STRIDE 72    // 64 + 8

#define WMMA_F16(A, B, C) \
  __builtin_amdgcn_wmma_f32_16x16x32_f16(false, (A), false, (B), (short)0, (C), false, false)

// A-fragment (16x32 f16) from a row-major f16 LDS tile, per the CDNA5 layout:
// lanes 0-15 : row=lane,    VGPR0-3 K=k0+0..7,  VGPR4-7 K=k0+16..23
// lanes 16-31: row=lane-16, VGPR0-3 K=k0+8..15, VGPR4-7 K=k0+24..31
__device__ __forceinline__ v16h a_frag_lds(const _Float16* tile, int stride,
                                           int k0, int lane) {
  const int row  = lane & 15;
  const int koff = k0 + ((lane >> 4) << 3);
  const _Float16* p = tile + row * stride + koff;
  v8h c0 = *(const v8h*)(p);        // 16B: K = koff .. koff+7
  v8h c1 = *(const v8h*)(p + 16);   // 16B: K = koff+16 .. koff+23
  return __builtin_shufflevector(c0, c1, 0,1,2,3,4,5,6,7,8,9,10,11,12,13,14,15);
}

// B-fragment (32x16 f16) gathered from a row-major fp32 [K x N] matrix.
// lane holds K = kbase+0..7 and kbase+16..23 of one column n (2 per VGPR).
__device__ __forceinline__ v16h b_frag_gather(const float* __restrict__ m,
                                              int ldm, int n, int kbase) {
  v16h r;
#pragma unroll
  for (int j = 0; j < 16; ++j)
    r[j] = (_Float16)m[(size_t)(kbase + (j < 8 ? j : j + 8)) * ldm + n];
  return r;
}

// tanh via the TRANS pipe (v_exp_f32 + fast rcp) — co-executes with XDL WMMA.
// Saturates correctly: x->+inf => 1, x->-inf => -1.
__device__ __forceinline__ float fast_tanh(float x) {
  float e = __expf(2.0f * x);
  return 1.0f - __fdividef(2.0f, e + 1.0f);
}

__global__ __launch_bounds__(256)
void rnn_scan_kernel(const float* __restrict__ x,   const float* __restrict__ w_h,
                     const float* __restrict__ u_h, const float* __restrict__ b_h,
                     const float* __restrict__ w_y, const float* __restrict__ b_y,
                     float* __restrict__ out_h,     float* __restrict__ out_y)
{
  __shared__ _Float16 s_h[2][16 * H_STRIDE];   // ping-pong hidden state (f16)
  __shared__ _Float16 s_x[2][16 * X_STRIDE];   // ping-pong x_t tile (f16)

  const int tid   = threadIdx.x;
  const int lane  = tid & 31;
  const int wid   = tid >> 5;                  // 8 waves
  const int r0    = blockIdx.x * 16;           // batch-row tile base
  const int nlo   = lane & 15;
  const int khi16 = (lane >> 4) << 3;          // B-frag: hi lanes take K+8

  // ---------- one-time: weight slices -> register fragments (reused 1024x) ----------
  const int n0 = wid * 32;                     // this wave's H columns [n0, n0+32)
  v16h Bu[2][8];                               // u_h: 2 N-tiles x 8 K-steps
#pragma unroll
  for (int nt = 0; nt < 2; ++nt)
#pragma unroll
    for (int ks = 0; ks < 8; ++ks)
      Bu[nt][ks] = b_frag_gather(u_h, RNN_H, n0 + nt * 16 + nlo, ks * 32 + khi16);

  v16h Bw[2][2];                               // w_h: 2 N-tiles x 2 K-steps (K=64)
#pragma unroll
  for (int nt = 0; nt < 2; ++nt)
#pragma unroll
    for (int ks = 0; ks < 2; ++ks)
      Bw[nt][ks] = b_frag_gather(w_h, RNN_H, n0 + nt * 16 + nlo, ks * 32 + khi16);

  v16h By[8];                                  // w_y: only waves 0,1 (O=32)
  float bias_y = 0.0f;
  if (wid < 2) {
#pragma unroll
    for (int ks = 0; ks < 8; ++ks)
      By[ks] = b_frag_gather(w_y, RNN_O, wid * 16 + nlo, ks * 32 + khi16);
    bias_y = b_y[wid * 16 + nlo];
  }
  const float bias_h0 = b_h[n0 + nlo];
  const float bias_h1 = b_h[n0 + 16 + nlo];

  // ---------- init: h0 = 0, stage x_0 ----------
  for (int i = tid; i < 16 * H_STRIDE; i += 256) s_h[0][i] = (_Float16)0.0f;

  const int xrow = tid >> 4;                   // 0..15
  const int xcol = (tid & 15) << 2;            // 0..60 step 4
  const float* xrow_base = x + (size_t)(r0 + xrow) * RNN_S * RNN_I + xcol;
  {
    v4f xv = *(const v4f*)(xrow_base);
    v4h xh;
#pragma unroll
    for (int e = 0; e < 4; ++e) xh[e] = (_Float16)xv[e];
    *(v4h*)(&s_x[0][xrow * X_STRIDE + xcol]) = xh;
  }
  __syncthreads();

  int cur = 0;
  for (int t = 0; t < RNN_S; ++t) {
    // Issue next-step x loads early; prefetch one more step ahead.
    v4f xnext = {0.0f, 0.0f, 0.0f, 0.0f};
    if (t + 1 < RNN_S) xnext = *(const v4f*)(xrow_base + (size_t)(t + 1) * RNN_I);
    if (t + 2 < RNN_S) __builtin_prefetch(xrow_base + (size_t)(t + 2) * RNN_I, 0, 1);

    // h_new[:, n0:n0+32] = tanh(b_h + x_t @ w_h + h @ u_h)
    v8f acc0, acc1;
#pragma unroll
    for (int e = 0; e < 8; ++e) { acc0[e] = bias_h0; acc1[e] = bias_h1; }

    const _Float16* px = s_x[t & 1];
    const _Float16* ph = s_h[cur];

    // 10 A-fragments: s=0,1 from x (K=64), s=2..9 from h (K=256).
    // 2-deep pipeline: frag s+2's ds loads are issued before frag s is consumed.
    v16h aP = a_frag_lds(px, X_STRIDE, 0, lane);
    v16h aQ = a_frag_lds(px, X_STRIDE, 32, lane);
#pragma unroll
    for (int s = 0; s < 10; ++s) {
      v16h a = aP;
      aP = aQ;
      if (s + 2 < 10) aQ = a_frag_lds(ph, H_STRIDE, s * 32, lane);  // frag s+2
      const v16h b0 = (s < 2) ? Bw[0][s] : Bu[0][s - 2];
      const v16h b1 = (s < 2) ? Bw[1][s] : Bu[1][s - 2];
      acc0 = WMMA_F16(a, b0, acc0);
      acc1 = WMMA_F16(a, b1, acc1);
    }

#pragma unroll
    for (int e = 0; e < 8; ++e) {
      acc0[e] = fast_tanh(acc0[e]);
      acc1[e] = fast_tanh(acc1[e]);
    }

    // Publish h_new (f16) into the other LDS buffer.
    {
      _Float16* dst = s_h[cur ^ 1];
#pragma unroll
      for (int e = 0; e < 8; ++e) {
        const int row = e + ((lane >> 4) << 3);
        dst[row * H_STRIDE + n0 + nlo]      = (_Float16)acc0[e];
        dst[row * H_STRIDE + n0 + 16 + nlo] = (_Float16)acc1[e];
      }
    }

    if (t == RNN_S - 1) {                      // h_final in fp32, from accumulators
#pragma unroll
      for (int e = 0; e < 8; ++e) {
        const int row = e + ((lane >> 4) << 3);
        out_h[(size_t)(r0 + row) * RNN_H + n0 + nlo]      = acc0[e];
        out_h[(size_t)(r0 + row) * RNN_H + n0 + 16 + nlo] = acc1[e];
      }
    }

    // Stage x_{t+1} into the other x buffer (before the same barrier).
    if (t + 1 < RNN_S) {
      v4h xh;
#pragma unroll
      for (int e = 0; e < 4; ++e) xh[e] = (_Float16)xnext[e];
      *(v4h*)(&s_x[(t + 1) & 1][xrow * X_STRIDE + xcol]) = xh;
    }

    __syncthreads();                           // one barrier per timestep

    // y_t = leaky_relu(h_new @ w_y + b_y); waves 0,1 own 16 output cols each.
    if (wid < 2) {
      const _Float16* phn = s_h[cur ^ 1];
      v8f accy;
#pragma unroll
      for (int e = 0; e < 8; ++e) accy[e] = bias_y;

      v16h yP = a_frag_lds(phn, H_STRIDE, 0, lane);
      v16h yQ = a_frag_lds(phn, H_STRIDE, 32, lane);
#pragma unroll
      for (int ks = 0; ks < 8; ++ks) {
        v16h a = yP;
        yP = yQ;
        if (ks + 2 < 8) yQ = a_frag_lds(phn, H_STRIDE, (ks + 2) * 32, lane);
        accy = WMMA_F16(a, By[ks], accy);
      }
#pragma unroll
      for (int e = 0; e < 8; ++e) {
        const int row = e + ((lane >> 4) << 3);
        float v = accy[e];
        v = fmaxf(v, 0.01f * v);               // leaky_relu, slope 0.01
        out_y[((size_t)(r0 + row) * RNN_S + t) * RNN_O + wid * 16 + nlo] = v;
      }
    }

    cur ^= 1;
  }
}

extern "C" void kernel_launch(void* const* d_in, const int* in_sizes, int n_in,
                              void* d_out, int out_size, void* d_ws, size_t ws_size,
                              hipStream_t stream) {
  const float* x   = (const float*)d_in[0];
  const float* w_h = (const float*)d_in[1];
  const float* u_h = (const float*)d_in[2];
  const float* b_h = (const float*)d_in[3];
  const float* w_y = (const float*)d_in[4];
  const float* b_y = (const float*)d_in[5];

  float* out_h = (float*)d_out;                          // [B, H]
  float* out_y = out_h + (size_t)RNN_B * RNN_H;          // [B, S, O]

  dim3 grid(RNN_B / 16);   // 128 blocks
  dim3 block(256);         // 8 wave32
  rnn_scan_kernel<<<grid, block, 0, stream>>>(x, w_h, u_h, b_h, w_y, b_y,
                                              out_h, out_y);
}
